// LSTMDecoder_14980845928456
// MI455X (gfx1250) — compile-verified, matrix operand order checked by reading, not run on hardware
//
#include <hip/hip_runtime.h>

// ---------------------------------------------------------------------------
// LSTM greedy decoder for MI455X (gfx1250, wave32, WMMA bf16 16x16x32).
// Persistent batch-tile kernel: each workgroup owns 32 batch rows, keeps
// h (bf16, ping-pong) and c (f32) in LDS, runs all 127 steps with no global
// sync.  Each wave owns BOTH 16-row blocks and a 64-unit hidden slice, so
// every W_hh B-fragment is loaded once and feeds two WMMAs (32 FLOP/L2-byte).
// tok@W_ih.T is a per-row column gather (tok is one-hot).
// ---------------------------------------------------------------------------

typedef __attribute__((ext_vector_type(16))) __bf16 bf16x16;
typedef __attribute__((ext_vector_type(8)))  float  f32x8;

#define HID    512
#define VOC    128
#define FOURH  2048
#define NSTEP  127      // SEQ_LEN - 1
#define TILE_B 32       // batch rows per workgroup
#define NTHR   256      // 8 waves

// ---- fast transcendentals (lower to v_exp_f32 / v_rcp_f32) ----------------
__device__ __forceinline__ float fast_exp(float x) {
  return __builtin_amdgcn_exp2f(x * 1.4426950408889634f);
}
__device__ __forceinline__ float fast_sigmoid(float x) {
  return __builtin_amdgcn_rcpf(1.0f + fast_exp(-x));
}
__device__ __forceinline__ float fast_tanh(float x) {
  x = fminf(fmaxf(x, -15.0f), 15.0f);          // avoid inf*0 -> NaN
  float e = fast_exp(-2.0f * x);
  return (1.0f - e) * __builtin_amdgcn_rcpf(1.0f + e);
}

// ---- WMMA fragment helpers -------------------------------------------------
// Per ISA 7.12.2 (16-bit A 16x32): lane L<16 holds row M=L, K = {k0..k0+7} u
// {k0+16..k0+23}; lane L+16 holds row M=L, K = {k0+8..} u {k0+24..}.
// B (32x16) mirrors with lane = column.  Both: two 16B loads at element
// offsets (k0 + 8*half) and (k0 + 8*half + 16).
__device__ __forceinline__ bf16x16 load_frag(const __bf16* p) {
  bf16x16 v;
  uint4* u = reinterpret_cast<uint4*>(&v);
  u[0] = *reinterpret_cast<const uint4*>(p);        // 8 bf16
  u[1] = *reinterpret_cast<const uint4*>(p + 16);   // next 8 bf16, +32 bytes
  return v;
}

__device__ __forceinline__ f32x8 wmma_bf16(bf16x16 a, bf16x16 b, f32x8 c) {
  return __builtin_amdgcn_wmma_f32_16x16x32_bf16(
      /*neg_a=*/false, a, /*neg_b=*/false, b,
      /*c_mod=*/(short)0, c, /*reuse_a=*/false, /*reuse_b=*/false);
}

// ---- one-time weight conversion kernels ------------------------------------
__global__ void cvt_bf16_kernel(const float* __restrict__ in,
                                __bf16* __restrict__ out, int n) {
  int i = blockIdx.x * NTHR + threadIdx.x;
  if (i < n) out[i] = (__bf16)in[i];
}

__global__ void bias_sum_kernel(const float* __restrict__ a,
                                const float* __restrict__ b,
                                float* __restrict__ o, int n) {
  int i = blockIdx.x * NTHR + threadIdx.x;
  if (i < n) o[i] = a[i] + b[i];
}

// ---- main persistent decoder kernel ----------------------------------------
__global__ __launch_bounds__(NTHR, 1)
void lstm_decoder_kernel(const float*  __restrict__ z,
                         const float*  __restrict__ W_ih,   // (2048, 128) f32
                         const float*  __restrict__ b_cz,   // (512,)
                         const float*  __restrict__ b_fc,   // (128,)
                         const __bf16* __restrict__ Whh,    // (2048, 512) bf16
                         const __bf16* __restrict__ Wcz,    // (512, 512)  bf16
                         const __bf16* __restrict__ Wfc,    // (128, 512)  bf16
                         const float*  __restrict__ bias,   // b_ih + b_hh (2048,)
                         float*        __restrict__ out) {  // (8192, 127, 128)
  extern __shared__ char smem[];
  __bf16* h0_lds = (__bf16*)smem;                                  // [32][512] 32KB
  __bf16* h1_lds = (__bf16*)(smem + TILE_B * HID * 2);             // [32][512] 32KB
  float*  c_lds  = (float*)(smem + 2 * TILE_B * HID * 2);          // [32][512] 64KB
  float*  l_lds  = (float*)(smem + 2 * TILE_B * HID * 2 + TILE_B * HID * 4); // [32][128]
  int*    tok    = (int*)(smem + 2 * TILE_B * HID * 2 + TILE_B * HID * 4 + TILE_B * VOC * 4);

  const int tid   = threadIdx.x;
  const int wave  = tid >> 5;        // 0..7 -> hidden slice [64*wave, 64*wave+64)
  const int lane  = tid & 31;
  const int half  = lane >> 4;       // lane half (K interleave select)
  const int m     = lane & 15;       // A: row within block / B: column / C,D: N
  const size_t blk = (size_t)blockIdx.x * TILE_B;

  // ---- stage h0 = z (bf16), tok0 = 0 ----
  const float* zt = z + blk * HID;
  for (int i = tid; i < TILE_B * HID; i += NTHR) h0_lds[i] = (__bf16)zt[i];
  if (tid < TILE_B) tok[tid] = 0;
  __syncthreads();

  // A-fragment base offsets for the two 16-row blocks (element index into h)
  const int a0off = m * HID + 8 * half;           // rows 0..15
  const int a1off = (16 + m) * HID + 8 * half;    // rows 16..31

  // ---- c0 = relu(z @ Wcz^T + b_cz): wave-exclusive 64-unit slice ----
  for (int jt = 0; jt < 4; ++jt) {
    const int jBase = wave * 64 + jt * 16;
    f32x8 c0a = {}, c0b = {};
#pragma unroll 1
    for (int kp = 0; kp < 2; ++kp) {
      bf16x16 a0[8], a1[8];
#pragma unroll
      for (int k = 0; k < 8; ++k) {
        const int ko = kp * 256 + k * 32;
        a0[k] = load_frag(&h0_lds[a0off + ko]);
        a1[k] = load_frag(&h0_lds[a1off + ko]);
      }
#pragma unroll
      for (int k = 0; k < 8; ++k) {
        const int ko = kp * 256 + k * 32 + 8 * half;
        bf16x16 b = load_frag(&Wcz[(size_t)(jBase + m) * HID + ko]);
        c0a = wmma_bf16(a0[k], b, c0a);
        c0b = wmma_bf16(a1[k], b, c0b);
      }
    }
    const int j = jBase + m;
    const float bb = b_cz[j];
#pragma unroll
    for (int r = 0; r < 8; ++r) {
      const int lr0 = r + 8 * half;
      c_lds[lr0 * HID + j]        = fmaxf(c0a[r] + bb, 0.0f);
      c_lds[(16 + lr0) * HID + j] = fmaxf(c0b[r] + bb, 0.0f);
    }
  }

  // ---- time loop: ping-pong h, 3 workgroup barriers per step ----
  for (int step = 0; step < NSTEP; ++step) {
    const __bf16* hc = (step & 1) ? h1_lds : h0_lds;   // current h (read)
    __bf16*       hn = (step & 1) ? h0_lds : h1_lds;   // next h (write)

    __syncthreads();   // (T) tok ready; prev step fully done with hn buffer

    // gates = h @ Whh^T + bias + W_ih[:, tok]; this wave: 32 rows x 64 hid,
    // all four gates -> LSTM cell entirely in registers.
    for (int jt = 0; jt < 4; ++jt) {
      const int jBase = wave * 64 + jt * 16;
      f32x8 gi0 = {}, gi1 = {}, gf0 = {}, gf1 = {};
      f32x8 gg0 = {}, gg1 = {}, go0 = {}, go1 = {};
#pragma unroll 1
      for (int kp = 0; kp < 2; ++kp) {
        bf16x16 a0[8], a1[8];
#pragma unroll
        for (int k = 0; k < 8; ++k) {
          const int ko = kp * 256 + k * 32;
          a0[k] = load_frag(&hc[a0off + ko]);
          a1[k] = load_frag(&hc[a1off + ko]);
        }
#pragma unroll
        for (int k = 0; k < 8; ++k) {
          const int ko = kp * 256 + k * 32 + 8 * half;
          bf16x16 bI = load_frag(&Whh[(size_t)(jBase + m) * HID + ko]);
          gi0 = wmma_bf16(a0[k], bI, gi0);
          gi1 = wmma_bf16(a1[k], bI, gi1);
          bf16x16 bF = load_frag(&Whh[(size_t)(512 + jBase + m) * HID + ko]);
          gf0 = wmma_bf16(a0[k], bF, gf0);
          gf1 = wmma_bf16(a1[k], bF, gf1);
          bf16x16 bG = load_frag(&Whh[(size_t)(1024 + jBase + m) * HID + ko]);
          gg0 = wmma_bf16(a0[k], bG, gg0);
          gg1 = wmma_bf16(a1[k], bG, gg1);
          bf16x16 bO = load_frag(&Whh[(size_t)(1536 + jBase + m) * HID + ko]);
          go0 = wmma_bf16(a0[k], bO, go0);
          go1 = wmma_bf16(a1[k], bO, go1);
        }
      }
      const int j = jBase + m;
      const float bi = bias[j],        bf = bias[512 + j];
      const float bg = bias[1024 + j], bo = bias[1536 + j];

      auto cell = [&](int lr, float xi, float xf, float xg, float xo) {
        const int tk = tok[lr];
        xi += bi + W_ih[(size_t)j * VOC + tk];
        xf += bf + W_ih[(size_t)(512 + j) * VOC + tk];
        xg += bg + W_ih[(size_t)(1024 + j) * VOC + tk];
        xo += bo + W_ih[(size_t)(1536 + j) * VOC + tk];
        float iv = fast_sigmoid(xi), fv = fast_sigmoid(xf);
        float gv = fast_tanh(xg),    ov = fast_sigmoid(xo);
        float cn = fv * c_lds[lr * HID + j] + iv * gv;
        c_lds[lr * HID + j] = cn;
        hn[lr * HID + j] = (__bf16)(ov * fast_tanh(cn));
      };
#pragma unroll
      for (int r = 0; r < 8; ++r) {
        const int lr0 = r + 8 * half;           // C/D layout: VGPR r -> rows r, r+8
        cell(lr0,      gi0[r], gf0[r], gg0[r], go0[r]);
        cell(16 + lr0, gi1[r], gf1[r], gg1[r], go1[r]);
      }
    }
    __syncthreads();   // (B) h_new complete across all hidden slices

    // logits = relu(h_new @ Wfc^T + b_fc): wave covers 32 rows x 16 vocab
    {
      const int vBase = wave * 16;
      f32x8 l0 = {}, l1 = {};
#pragma unroll 1
      for (int kp = 0; kp < 2; ++kp) {
        bf16x16 a0[8], a1[8];
#pragma unroll
        for (int k = 0; k < 8; ++k) {
          const int ko = kp * 256 + k * 32;
          a0[k] = load_frag(&hn[a0off + ko]);
          a1[k] = load_frag(&hn[a1off + ko]);
        }
#pragma unroll
        for (int k = 0; k < 8; ++k) {
          const int ko = kp * 256 + k * 32 + 8 * half;
          bf16x16 b = load_frag(&Wfc[(size_t)(vBase + m) * HID + ko]);
          l0 = wmma_bf16(a0[k], b, l0);
          l1 = wmma_bf16(a1[k], b, l1);
        }
      }
      const int v = vBase + m;
      const float bb = b_fc[v];
#pragma unroll
      for (int r = 0; r < 8; ++r) {
        const int lr0 = r + 8 * half;
        float x0 = fmaxf(l0[r] + bb, 0.0f);
        float x1 = fmaxf(l1[r] + bb, 0.0f);
        l_lds[lr0 * VOC + v]        = x0;
        l_lds[(16 + lr0) * VOC + v] = x1;
        out[(blk + lr0) * (size_t)(NSTEP * VOC) + (size_t)step * VOC + v]      = x0;
        out[(blk + 16 + lr0) * (size_t)(NSTEP * VOC) + (size_t)step * VOC + v] = x1;
      }
    }
    __syncthreads();   // (C) logits visible in LDS

    // greedy argmax, all 8 waves: 8 lanes per row, first-max semantics
    // (in-order chunk scan, ties in the xor-reduction pick the lower index).
    {
      const int sub = lane & 7;        // contiguous 16-col chunk per lane
      const int row = wave * 4 + (lane >> 3);
      const float4* p4 = (const float4*)&l_lds[row * VOC + sub * 16];
      float4 q0 = p4[0], q1 = p4[1], q2 = p4[2], q3 = p4[3];
      float vv[16] = {q0.x, q0.y, q0.z, q0.w, q1.x, q1.y, q1.z, q1.w,
                      q2.x, q2.y, q2.z, q2.w, q3.x, q3.y, q3.z, q3.w};
      float best = vv[0];
      int   bidx = sub * 16;
#pragma unroll
      for (int i = 1; i < 16; ++i)
        if (vv[i] > best) { best = vv[i]; bidx = sub * 16 + i; }
#pragma unroll
      for (int off = 1; off < 8; off <<= 1) {
        float ob = __shfl_xor(best, off, 32);
        int   oi = __shfl_xor(bidx, off, 32);
        if (ob > best || (ob == best && oi < bidx)) { best = ob; bidx = oi; }
      }
      if (sub == 0) tok[row] = bidx;
    }
    // barrier (T) at top of next iteration publishes tok
  }
}

// ---------------------------------------------------------------------------
extern "C" void kernel_launch(void* const* d_in, const int* in_sizes, int n_in,
                              void* d_out, int out_size, void* d_ws, size_t ws_size,
                              hipStream_t stream) {
  const float* z    = (const float*)d_in[0];
  const float* W_ih = (const float*)d_in[1];
  const float* W_hh = (const float*)d_in[2];
  const float* b_ih = (const float*)d_in[3];
  const float* b_hh = (const float*)d_in[4];
  const float* W_cz = (const float*)d_in[5];
  const float* b_cz = (const float*)d_in[6];
  const float* W_fc = (const float*)d_in[7];
  const float* b_fc = (const float*)d_in[8];
  float* out = (float*)d_out;

  // workspace layout (bf16 weights + fused bias), recomputed every call
  char* ws = (char*)d_ws;
  __bf16* whh_bf = (__bf16*)ws;                                   // 2048*512*2 = 2 MB
  __bf16* wcz_bf = (__bf16*)(ws + (size_t)FOURH * HID * 2);       // 512*512*2
  __bf16* wfc_bf = (__bf16*)(ws + (size_t)FOURH * HID * 2 + (size_t)HID * HID * 2);
  float*  biasf  = (float*)(ws + (size_t)FOURH * HID * 2 + (size_t)HID * HID * 2
                               + (size_t)VOC * HID * 2);          // 2048 f32

  const int n_whh = FOURH * HID;   // 1048576
  const int n_wcz = HID * HID;     // 262144
  const int n_wfc = VOC * HID;     // 65536

  cvt_bf16_kernel<<<(n_whh + NTHR - 1) / NTHR, NTHR, 0, stream>>>(W_hh, whh_bf, n_whh);
  cvt_bf16_kernel<<<(n_wcz + NTHR - 1) / NTHR, NTHR, 0, stream>>>(W_cz, wcz_bf, n_wcz);
  cvt_bf16_kernel<<<(n_wfc + NTHR - 1) / NTHR, NTHR, 0, stream>>>(W_fc, wfc_bf, n_wfc);
  bias_sum_kernel<<<(FOURH + NTHR - 1) / NTHR, NTHR, 0, stream>>>(b_ih, b_hh, biasf, FOURH);

  const int smem_bytes = 2 * TILE_B * HID * 2   // h ping-pong (bf16)
                       + TILE_B * HID * 4       // c (f32)
                       + TILE_B * VOC * 4       // logits (f32)
                       + TILE_B * 4;            // tok indices
  const int nblocks = 8192 / TILE_B;            // 256 persistent workgroups
  lstm_decoder_kernel<<<nblocks, NTHR, smem_bytes, stream>>>(
      z, W_ih, b_cz, b_fc, whh_bf, wcz_bf, wfc_bf, biasf, out);
}
